// calc_ssl_loss_v3_83279415869553
// MI455X (gfx1250) — matrix operand order repaired and checked by reading, not compiled, not run on hardware
//
#include <hip/hip_runtime.h>
#include <math.h>

#define D 128
#define SSL_REG_F 0.1f
// log2(e) / SSL_TEMP  =  1.4426950408889634 / 0.2
#define EXP2_SCALE 7.2134752044448170f
#define INV_TEMP 5.0f
#define LN2_F 0.69314718055994530942f

// padded column stride in halves: 272 B = 68 dwords -> 4-bank rotation
#define BCOL_STRIDE 136
// one 16-col sub-tile in halves
#define BTILE_H (16 * BCOL_STRIDE)
// sub-tiles (16 cols each) per staged iteration
#define NQ 4

typedef _Float16 v16h __attribute__((ext_vector_type(16)));
typedef _Float16 v8h  __attribute__((ext_vector_type(8)));
typedef _Float16 v4h  __attribute__((ext_vector_type(4)));
typedef float    v8f  __attribute__((ext_vector_type(8)));

union AFrag { v16h v; v8h h[2]; };

// ---------------------------------------------------------------------------
// Phase 1: gather + row L2-normalize (f32 -> f16) + diagonal cosine term.
// One wave (32 lanes) per row; each lane handles 4 contiguous columns.
// ---------------------------------------------------------------------------
__global__ void ssl_normalize_kernel(const float* __restrict__ ua1,
                                     const float* __restrict__ ua2,
                                     const float* __restrict__ ia1,
                                     const float* __restrict__ ia2,
                                     const int*  __restrict__ users,
                                     const int*  __restrict__ items,
                                     _Float16* __restrict__ Z1,
                                     _Float16* __restrict__ Z2,
                                     float* __restrict__ diag,
                                     int B)
{
    const int wave = threadIdx.x >> 5;
    const int lane = threadIdx.x & 31;
    const int row  = blockIdx.x * 8 + wave;
    const int N    = 2 * B;
    if (row >= N) return;

    const float* e1;
    const float* e2;
    if (row < B) {
        const long long g = (long long)users[row];
        e1 = ua1 + g * D;
        e2 = ua2 + g * D;
    } else {
        const long long g = (long long)items[row - B];
        e1 = ia1 + g * D;
        e2 = ia2 + g * D;
    }

    const float4 a = ((const float4*)e1)[lane];
    const float4 b = ((const float4*)e2)[lane];

    float s11 = a.x*a.x + a.y*a.y + a.z*a.z + a.w*a.w;
    float s22 = b.x*b.x + b.y*b.y + b.z*b.z + b.w*b.w;
    float s12 = a.x*b.x + a.y*b.y + a.z*b.z + a.w*b.w;

    #pragma unroll
    for (int m = 1; m < 32; m <<= 1) {
        s11 += __shfl_xor(s11, m, 32);
        s22 += __shfl_xor(s22, m, 32);
        s12 += __shfl_xor(s12, m, 32);
    }

    const float rn1 = rsqrtf(s11);
    const float rn2 = rsqrtf(s22);

    v4h z1o, z2o;
    z1o[0] = (_Float16)(a.x * rn1); z1o[1] = (_Float16)(a.y * rn1);
    z1o[2] = (_Float16)(a.z * rn1); z1o[3] = (_Float16)(a.w * rn1);
    z2o[0] = (_Float16)(b.x * rn2); z2o[1] = (_Float16)(b.y * rn2);
    z2o[2] = (_Float16)(b.z * rn2); z2o[3] = (_Float16)(b.w * rn2);

    ((v4h*)(Z1 + (long long)row * D))[lane] = z1o;
    ((v4h*)(Z2 + (long long)row * D))[lane] = z2o;

    if (lane == 0) diag[row] = s12 * rn1 * rn2;   // cos(z1_i, z2_i)
}

// ---------------------------------------------------------------------------
// Phase 2: fused  rowsum_i = sum_j exp( (Z1 Z2^T)_ij / temp )
// Block = 256 threads (8 waves). Wave w owns rows [base+16w, base+16w+16).
// Per iteration a 64-column B slab (4 x 16-col sub-tiles) is staged in
// double-buffered padded LDS; each wave runs 4 INDEPENDENT 4-deep
// v_wmma_f32_16x16x32_f16 chains (one per sub-tile) off the same register-
// resident A fragment, so the scheduler always has 4 dependency-free WMMAs
// to issue back-to-back. One barrier per 64 columns; last iteration peeled
// so the steady-state loop has no branches.
// ---------------------------------------------------------------------------
__global__ void ssl_gemm_exp_rowsum_kernel(const _Float16* __restrict__ Z1,
                                           const _Float16* __restrict__ Z2,
                                           float* __restrict__ rowsum,
                                           int N)
{
    // [2 buffers][NQ sub-tiles][16 cols x 136 halves]  = 34816 B
    __shared__ __align__(16) _Float16 ldsB[2 * NQ * BTILE_H];

    const int tid  = threadIdx.x;
    const int wave = tid >> 5;
    const int lane = tid & 31;
    const int m16  = lane & 15;   // row-in-tile (A) / column-in-tile (B)
    const int h    = lane >> 4;   // lane half selects K sub-range (contiguous k = 16h + s)

    const int rowbase = blockIdx.x * 128 + wave * 16;

    // staging role: 16 B of sub-tile column `scol`, part `spart`, for each q
    const int scol  = tid >> 4;   // 0..15
    const int spart = tid & 15;   // 0..15

    // --- preload A fragments: contiguous-K 16x32 layout (k = 16h + s), 4 K-chunks ---
    // h==0 lanes: K = 32c+[0..15]
    // h==1 lanes: K = 32c+[16..31]
    AFrag a[4];
    {
        const _Float16* arow = Z1 + (long long)(rowbase + m16) * D;
        #pragma unroll
        for (int c = 0; c < 4; ++c) {
            a[c].h[0] = *(const v8h*)(arow + 32 * c + 16 * h);
            a[c].h[1] = *(const v8h*)(arow + 32 * c + 16 * h + 8);
        }
    }

    float acc[8];
    #pragma unroll
    for (int r = 0; r < 8; ++r) acc[r] = 0.0f;

    // per-wave LDS read base within a sub-tile
    const int brd = m16 * BCOL_STRIDE + 16 * h;

    const int niter = N >> 6;     // 64 columns per iteration

    // prologue: stage slab 0 into buffer 0
    #pragma unroll
    for (int q = 0; q < NQ; ++q) {
        const uint4 ld0 =
            ((const uint4*)(Z2 + (long long)(q * 16 + scol) * D))[spart];
        *(uint4*)&ldsB[q * BTILE_H + scol * BCOL_STRIDE + spart * 8] = ld0;
    }
    __syncthreads();

    for (int j = 0; j + 1 < niter; ++j) {
        const int cur = (j & 1) * (NQ * BTILE_H);
        const int nxt = ((j + 1) & 1) * (NQ * BTILE_H);

        // issue next slab's global loads FIRST; latency overlaps the WMMAs
        uint4 nld[NQ];
        #pragma unroll
        for (int q = 0; q < NQ; ++q)
            nld[q] = ((const uint4*)(Z2 +
                (long long)((j + 1) * 64 + q * 16 + scol) * D))[spart];

        // 4 independent accumulator chains, interleaved per K step
        v8f cacc[NQ];
        #pragma unroll
        for (int q = 0; q < NQ; ++q) cacc[q] = (v8f){};
        #pragma unroll
        for (int c = 0; c < 4; ++c) {
            #pragma unroll
            for (int q = 0; q < NQ; ++q) {
                const v16h b = *(const v16h*)
                    &ldsB[cur + q * BTILE_H + brd + 32 * c];
                cacc[q] = __builtin_amdgcn_wmma_f32_16x16x32_f16(
                    false, a[c].v, false, b, (short)0, cacc[q], false, false);
            }
        }

        // fused epilogue: exp(x/temp) = exp2(x*log2e/temp); cosines in [-1,1]
        // -> arg in [-7.22,7.22], raw v_exp_f32 is safe (no clamp path).
        #pragma unroll
        for (int q = 0; q < NQ; ++q)
            #pragma unroll
            for (int r = 0; r < 8; ++r)
                acc[r] += __builtin_amdgcn_exp2f(cacc[q][r] * EXP2_SCALE);

        // stage next slab into the other buffer (previous readers of that
        // buffer were fenced by the end-of-iteration barrier of j-1)
        #pragma unroll
        for (int q = 0; q < NQ; ++q)
            *(uint4*)&ldsB[nxt + q * BTILE_H + scol * BCOL_STRIDE + spart * 8]
                = nld[q];

        __syncthreads();
    }

    // peeled final slab: compute only
    {
        const int cur = ((niter - 1) & 1) * (NQ * BTILE_H);
        v8f cacc[NQ];
        #pragma unroll
        for (int q = 0; q < NQ; ++q) cacc[q] = (v8f){};
        #pragma unroll
        for (int c = 0; c < 4; ++c) {
            #pragma unroll
            for (int q = 0; q < NQ; ++q) {
                const v16h b = *(const v16h*)
                    &ldsB[cur + q * BTILE_H + brd + 32 * c];
                cacc[q] = __builtin_amdgcn_wmma_f32_16x16x32_f16(
                    false, a[c].v, false, b, (short)0, cacc[q], false, false);
            }
        }
        #pragma unroll
        for (int q = 0; q < NQ; ++q)
            #pragma unroll
            for (int r = 0; r < 8; ++r)
                acc[r] += __builtin_amdgcn_exp2f(cacc[q][r] * EXP2_SCALE);
    }

    // C layout: VGPR r, lanes 0-15 -> row r, lanes 16-31 -> row r+8.
    // Reduce across the 16 lanes holding the same row (xor masks < 16).
    #pragma unroll
    for (int r = 0; r < 8; ++r) {
        float v = acc[r];
        v += __shfl_xor(v, 1, 32);
        v += __shfl_xor(v, 2, 32);
        v += __shfl_xor(v, 4, 32);
        v += __shfl_xor(v, 8, 32);
        acc[r] = v;
    }
    if (m16 == 0) {
        #pragma unroll
        for (int r = 0; r < 8; ++r)
            rowsum[rowbase + h * 8 + r] = acc[r];
    }
}

// ---------------------------------------------------------------------------
// Phase 3: loss = SSL_REG * sum_i ( log(rowsum_i) - diag_i / temp )
// rowsum_i strictly positive and in f32 range -> raw v_log_f32 safe.
// ---------------------------------------------------------------------------
__global__ void ssl_loss_reduce_kernel(const float* __restrict__ rowsum,
                                       const float* __restrict__ diag,
                                       float* __restrict__ out,
                                       int N)
{
    __shared__ float red[256];
    float s = 0.0f;
    for (int i = threadIdx.x; i < N; i += 256)
        s += __builtin_amdgcn_logf(rowsum[i]) * LN2_F - diag[i] * INV_TEMP;
    red[threadIdx.x] = s;
    __syncthreads();
    #pragma unroll
    for (int k = 128; k > 0; k >>= 1) {
        if (threadIdx.x < k) red[threadIdx.x] += red[threadIdx.x + k];
        __syncthreads();
    }
    if (threadIdx.x == 0) out[0] = SSL_REG_F * red[0];
}

// ---------------------------------------------------------------------------
extern "C" void kernel_launch(void* const* d_in, const int* in_sizes, int n_in,
                              void* d_out, int out_size, void* d_ws, size_t ws_size,
                              hipStream_t stream)
{
    const float* ua1 = (const float*)d_in[0];
    const float* ua2 = (const float*)d_in[1];
    const float* ia1 = (const float*)d_in[2];
    const float* ia2 = (const float*)d_in[3];
    const int* users = (const int*)d_in[4];
    const int* items = (const int*)d_in[5];

    const int B = in_sizes[4];      // 8192
    const int N = 2 * B;            // 16384

    // workspace layout: Z1 (N*128 f16) | Z2 (N*128 f16) | diag (N f32) | rowsum (N f32)
    char* ws = (char*)d_ws;
    _Float16* Z1 = (_Float16*)ws;
    _Float16* Z2 = Z1 + (size_t)N * D;
    float* diag   = (float*)(ws + 2 * (size_t)N * D * sizeof(_Float16));
    float* rowsum = diag + N;

    ssl_normalize_kernel<<<(N + 7) / 8, 256, 0, stream>>>(
        ua1, ua2, ia1, ia2, users, items, Z1, Z2, diag, B);

    ssl_gemm_exp_rowsum_kernel<<<N / 128, 256, 0, stream>>>(Z1, Z2, rowsum, N);

    ssl_loss_reduce_kernel<<<1, 256, 0, stream>>>(rowsum, diag, (float*)d_out, N);
}
